// PAM_Module_76459007803613
// MI455X (gfx1250) — matrix operand order repaired and checked by reading, not compiled, not run on hardware
//
#include <hip/hip_runtime.h>

typedef __attribute__((ext_vector_type(16))) _Float16 v16h;
typedef __attribute__((ext_vector_type(8)))  float    v8f;

constexpr int B  = 4;
constexpr int C  = 512;
constexpr int N  = 4096;   // H*W
constexpr int CR = 64;     // C/8
constexpr int MB = 32;     // query rows per workgroup

// LDS row pitches (padded to break bank conflicts)
constexpr int QLD = 72;    // halves
constexpr int PLD = 72;    // halves
constexpr int SLD = 65;    // floats

// ---------------------------------------------------------------------------
// WMMA fragment loaders (layouts from CDNA5 ISA, wave32)
// A 16x32 f16:  lane = half*16 + row; VGPR r   -> K = 16*half + 2r (contiguous k order chosen here, not the ISA table order)
// B 32x16 f16:  lane = half*16 + col; VGPR r   -> K = 16*half + 2r (same contiguous order as A, not the ISA table order)
// C/D 16x16 f32: VGPR g holds row (g + 8*half), col = lane&15
// ---------------------------------------------------------------------------
__device__ inline v16h lds_a_frag(const _Float16* s, int ld, int lane, int kbase) {
  int row = lane & 15, half = lane >> 4;
  v16h a;
#pragma unroll
  for (int r = 0; r < 8; ++r) {
    int k = kbase + 16 * half + 2 * r;
    a[2 * r]     = s[row * ld + k];
    a[2 * r + 1] = s[row * ld + k + 1];
  }
  return a;
}

__device__ inline v16h glb_b_frag_cm(const _Float16* g, int ld, int lane, int kbase) {
  // B[kk][col] = g[col*ld + kk]  (kk contiguous -> merges into global_load_b128)
  int col = lane & 15, half = lane >> 4;
  v16h bf;
#pragma unroll
  for (int r = 0; r < 8; ++r) {
    int kk = kbase + 16 * half + 2 * r;
    bf[2 * r]     = g[col * ld + kk];
    bf[2 * r + 1] = g[col * ld + kk + 1];
  }
  return bf;
}

// ---------------------------------------------------------------------------
// Projection: q[b,n,r] and kT[b,n,r] (f16, both j-major so fragment loads vectorize)
// ---------------------------------------------------------------------------
__global__ __launch_bounds__(256) void proj_qk_kernel(
    const float* __restrict__ x, const float* __restrict__ Wq, const float* __restrict__ bq,
    const float* __restrict__ Wk, const float* __restrict__ bk,
    _Float16* __restrict__ q16, _Float16* __restrict__ kT) {
  __shared__ float sWq[64][64];
  __shared__ float sWk[64][64];
  int tid = threadIdx.x;
  int tn = tid & 63, tg = tid >> 6;          // 64 pixels x 4 groups of 16 outputs
  int n = blockIdx.x * 64 + tn;
  int b = blockIdx.y;
  float accq[16], acck[16];
#pragma unroll
  for (int i = 0; i < 16; ++i) { accq[i] = 0.f; acck[i] = 0.f; }

  for (int cc = 0; cc < C; cc += 64) {
#pragma unroll
    for (int t = 0; t < 16; ++t) {
      int idx = t * 256 + tid;
      int r = idx >> 6, c = idx & 63;
      sWq[r][c] = Wq[r * C + cc + c];
      sWk[r][c] = Wk[r * C + cc + c];
    }
    __syncthreads();
    for (int c = 0; c < 64; ++c) {
      float xv = x[(size_t)((b * C + cc + c) * N) + n];
#pragma unroll
      for (int i = 0; i < 16; ++i) {
        accq[i] = fmaf(sWq[tg * 16 + i][c], xv, accq[i]);
        acck[i] = fmaf(sWk[tg * 16 + i][c], xv, acck[i]);
      }
    }
    __syncthreads();
  }
#pragma unroll
  for (int i = 0; i < 16; ++i) {
    int r = tg * 16 + i;
    q16[(size_t)(b * N + n) * CR + r] = (_Float16)(accq[i] + bq[r]);
    kT[(size_t)(b * N + n) * CR + r]  = (_Float16)(acck[i] + bk[r]);
  }
}

// ---------------------------------------------------------------------------
// Projection: v[b,c,n] (f16)
// ---------------------------------------------------------------------------
__global__ __launch_bounds__(256) void proj_v_kernel(
    const float* __restrict__ x, const float* __restrict__ Wv, const float* __restrict__ bv,
    _Float16* __restrict__ v16) {
  __shared__ float sWv[64][64];
  int tid = threadIdx.x;
  int tn = tid & 63, tg = tid >> 6;
  int n   = blockIdx.x * 64 + tn;
  int cg0 = blockIdx.y * 64;
  int b   = blockIdx.z;
  float acc[16];
#pragma unroll
  for (int i = 0; i < 16; ++i) acc[i] = 0.f;

  for (int cc = 0; cc < C; cc += 64) {
#pragma unroll
    for (int t = 0; t < 16; ++t) {
      int idx = t * 256 + tid;
      int r = idx >> 6, c = idx & 63;
      sWv[r][c] = Wv[(size_t)(cg0 + r) * C + cc + c];
    }
    __syncthreads();
    for (int c = 0; c < 64; ++c) {
      float xv = x[(size_t)((b * C + cc + c) * N) + n];
#pragma unroll
      for (int i = 0; i < 16; ++i)
        acc[i] = fmaf(sWv[tg * 16 + i][c], xv, acc[i]);
    }
    __syncthreads();
  }
#pragma unroll
  for (int i = 0; i < 16; ++i) {
    int cch = cg0 + tg * 16 + i;
    v16[(size_t)(b * C + cch) * N + n] = (_Float16)(acc[i] + bv[cch]);
  }
}

// ---------------------------------------------------------------------------
// Fused flash attention + out = V * A^T, one workgroup per (batch, 32 rows).
// Per 64-key iteration (no divergent branches; EXEC full at every WMMA):
//   every wave : one 16x16 score tile (rb = wave>>2, t = wave&3), 2 WMMAs
//   every wave : online softmax for 4 rows (32-lane shfl reductions)
//   every wave : rescale, O[2][4] += P(32x64) x V^T(64x64ch)  (16 WMMAs)
// ---------------------------------------------------------------------------
__global__ __launch_bounds__(256) void pam_attn_kernel(
    const _Float16* __restrict__ q16, const _Float16* __restrict__ kT,
    const _Float16* __restrict__ v16, const float* __restrict__ gamma_p,
    float* __restrict__ out) {
  __shared__ _Float16 sQ[MB * QLD];
  __shared__ float    sS[MB * SLD];
  __shared__ _Float16 sP[MB * PLD];
  __shared__ float    sScale[MB];
  __shared__ float    sL[MB];

  int tid  = threadIdx.x;
  int lane = tid & 31, wave = tid >> 5;
  int b  = blockIdx.y;
  int i0 = blockIdx.x * MB;
  int half = lane >> 4, col = lane & 15;
  float gm = gamma_p[0];
  const int c0   = wave * 64;        // this wave's output-channel slice
  const int rb_s = wave >> 2;        // score tile: row-block (0/1)
  const int t_s  = wave & 3;         // score tile: 16-col tile (0..3)

  // stage Q tile (32 x 64) into LDS
  for (int idx = tid; idx < MB * CR; idx += 256) {
    int m = idx >> 6, r = idx & 63;
    sQ[m * QLD + r] = q16[(size_t)(b * N + i0) * CR + idx];
  }
  __syncthreads();

  // Q fragments for this wave's score row-block
  v16h aQ0 = lds_a_frag(sQ + rb_s * 16 * QLD, QLD, lane, 0);
  v16h aQ1 = lds_a_frag(sQ + rb_s * 16 * QLD, QLD, lane, 32);

  v8f O[2][4];
#pragma unroll
  for (int rb = 0; rb < 2; ++rb)
#pragma unroll
    for (int t = 0; t < 4; ++t) O[rb][t] = v8f{};

  float mr[4], lr[4];
#pragma unroll
  for (int k = 0; k < 4; ++k) { mr[k] = -3.0e38f; lr[k] = 0.f; }

  for (int j0 = 0; j0 < N; j0 += 64) {
    // ---- score tile: S[rb_s*16.., t_s*16..] = Q x K^T -------------------
    {
      const _Float16* kb = kT + (size_t)(b * N + j0 + t_s * 16) * CR;
      v16h b0 = glb_b_frag_cm(kb, CR, lane, 0);
      v16h b1 = glb_b_frag_cm(kb, CR, lane, 32);
      v8f s{};
      s = __builtin_amdgcn_wmma_f32_16x16x32_f16(false, aQ0, false, b0, (short)0, s, false, false);
      s = __builtin_amdgcn_wmma_f32_16x16x32_f16(false, aQ1, false, b1, (short)0, s, false, false);
#pragma unroll
      for (int g = 0; g < 8; ++g)
        sS[(rb_s * 16 + g + 8 * half) * SLD + t_s * 16 + col] = s[g];
    }
    __syncthreads();

    // ---- distributed online softmax: 4 rows per wave --------------------
#pragma unroll
    for (int k = 0; k < 4; ++k) {
      int row = 4 * wave + k;
      float va = sS[row * SLD + lane], vb = sS[row * SLD + lane + 32];
      float rm = fmaxf(va, vb);
#pragma unroll
      for (int off = 1; off < 32; off <<= 1) rm = fmaxf(rm, __shfl_xor(rm, off));
      float mn = fmaxf(mr[k], rm);
      float sc = __expf(mr[k] - mn);
      float pa = __expf(va - mn), pb = __expf(vb - mn);
      float rs = pa + pb;
#pragma unroll
      for (int off = 1; off < 32; off <<= 1) rs += __shfl_xor(rs, off);
      lr[k] = lr[k] * sc + rs;
      mr[k] = mn;
      sP[row * PLD + lane]      = (_Float16)pa;
      sP[row * PLD + lane + 32] = (_Float16)pb;
      if (lane == 0) sScale[row] = sc;
    }
    __syncthreads();

    // ---- O += P x V^T  (per-wave 64-channel slice, 2 row-blocks) ---------
    float scl[2][8];
#pragma unroll
    for (int rb = 0; rb < 2; ++rb)
#pragma unroll
      for (int g = 0; g < 8; ++g) scl[rb][g] = sScale[rb * 16 + g + 8 * half];
#pragma unroll
    for (int rb = 0; rb < 2; ++rb)
#pragma unroll
      for (int t = 0; t < 4; ++t)
#pragma unroll
        for (int g = 0; g < 8; ++g) O[rb][t][g] *= scl[rb][g];

    v16h aP[2][2];
#pragma unroll
    for (int rb = 0; rb < 2; ++rb) {
      aP[rb][0] = lds_a_frag(sP + rb * 16 * PLD, PLD, lane, 0);
      aP[rb][1] = lds_a_frag(sP + rb * 16 * PLD, PLD, lane, 32);
    }

    if (j0 + 64 < N) {  // prefetch next tiles (global_prefetch_b8)
      __builtin_prefetch(v16 + (size_t)(b * C + c0) * N + j0 + 64, 0, 0);
      __builtin_prefetch(kT + (size_t)(b * N + j0 + 64) * CR, 0, 0);
    }

#pragma unroll
    for (int t = 0; t < 4; ++t) {
      const _Float16* vb = v16 + (size_t)(b * C + c0 + t * 16) * N + j0;
      v16h b0 = glb_b_frag_cm(vb, N, lane, 0);
      v16h b1 = glb_b_frag_cm(vb, N, lane, 32);
#pragma unroll
      for (int rb = 0; rb < 2; ++rb) {   // V fragments reused across row-blocks
        O[rb][t] = __builtin_amdgcn_wmma_f32_16x16x32_f16(false, aP[rb][0], false, b0, (short)0, O[rb][t], false, false);
        O[rb][t] = __builtin_amdgcn_wmma_f32_16x16x32_f16(false, aP[rb][1], false, b1, (short)0, O[rb][t], false, false);
      }
    }
    __syncthreads();  // protect sS/sP/sScale for next iteration
  }

  if (lane == 0) {
#pragma unroll
    for (int k = 0; k < 4; ++k) sL[4 * wave + k] = lr[k];
  }
  __syncthreads();

  float inv[2][8];
#pragma unroll
  for (int rb = 0; rb < 2; ++rb)
#pragma unroll
    for (int g = 0; g < 8; ++g) inv[rb][g] = gm / sL[rb * 16 + g + 8 * half];
#pragma unroll
  for (int rb = 0; rb < 2; ++rb)
#pragma unroll
    for (int t = 0; t < 4; ++t) {
      int c = c0 + t * 16 + col;
#pragma unroll
      for (int g = 0; g < 8; ++g) {
        int i = i0 + rb * 16 + g + 8 * half;
        out[(size_t)(b * C + c) * N + i] = O[rb][t][g] * inv[rb][g];
      }
    }
}

// ---------------------------------------------------------------------------
extern "C" void kernel_launch(void* const* d_in, const int* in_sizes, int n_in,
                              void* d_out, int out_size, void* d_ws, size_t ws_size,
                              hipStream_t stream) {
  const float* x  = (const float*)d_in[0];
  const float* Wq = (const float*)d_in[1];
  const float* bq = (const float*)d_in[2];
  const float* Wk = (const float*)d_in[3];
  const float* bk = (const float*)d_in[4];
  const float* Wv = (const float*)d_in[5];
  const float* bv = (const float*)d_in[6];
  const float* gm = (const float*)d_in[7];
  float* out = (float*)d_out;

  // workspace: q16 (2 MB) | kT (2 MB) | v16 (16.8 MB)
  _Float16* q16 = (_Float16*)d_ws;
  _Float16* kT  = q16 + (size_t)B * N * CR;
  _Float16* v16 = kT + (size_t)B * N * CR;

  proj_qk_kernel<<<dim3(N / 64, B), 256, 0, stream>>>(x, Wq, bq, Wk, bk, q16, kT);
  proj_v_kernel<<<dim3(N / 64, C / 64, B), 256, 0, stream>>>(x, Wv, bv, v16);
  pam_attn_kernel<<<dim3(N / MB, B), 256, 0, stream>>>(q16, kT, v16, gm, out);
}
